// ReformerTimeSeries_86474871538643
// MI455X (gfx1250) — matrix operand order repaired and checked by reading, not compiled, not run on hardware
//
#include <hip/hip_runtime.h>
#include <math.h>

// ---------------- model dims ----------------
#define D_MODEL 512
#define SEQ 512
#define BATCH 16
#define HEADS 8
#define DH 64
#define NH 4        // hash rounds
#define FFD 2048
#define MROWS (BATCH*SEQ)   // 8192
#define BHN (BATCH*HEADS)   // 128
#define NS (NH*SEQ)         // 2048 sorted length per (b,h)
#define CHUNK 256
#define NCH 8               // NH * N_BUCKETS chunks

typedef __attribute__((ext_vector_type(16))) __bf16 v16bf;
typedef __attribute__((ext_vector_type(8)))  float  v8f;
typedef __attribute__((ext_vector_type(4)))  unsigned int v4u;

union BF16Frag { v16bf v; v4u q[2]; };

typedef __attribute__((address_space(1))) int gas_i32;
typedef __attribute__((address_space(3))) int las_i32;

#if defined(__has_builtin)
#  if __has_builtin(__builtin_amdgcn_global_load_async_to_lds_b32)
#    define USE_ASYNC_LDS 1
#  endif
#endif
#ifndef USE_ASYNC_LDS
#  define USE_ASYNC_LDS 0
#endif

__device__ __forceinline__ unsigned short f2bf(float f) {
  unsigned int u = __float_as_uint(f);
  u += 0x7FFFu + ((u >> 16) & 1u);          // round-to-nearest-even
  return (unsigned short)(u >> 16);
}

// ---------------- weight f32 -> bf16, transposed to N x K ----------------
__global__ void f2bf_t_kernel(const float* __restrict__ src, unsigned short* __restrict__ dst,
                              int K, int N, int layers) {
  int idx = blockIdx.x * 256 + threadIdx.x;
  if (idx >= layers * K * N) return;
  int l = idx / (K * N);
  int rem = idx - l * (K * N);
  int k = rem / N, n = rem - k * N;
  dst[(size_t)l * K * N + (size_t)n * K + k] = f2bf(src[idx]);
}

// ---------------- embedding + positional ----------------
__global__ void embed_kernel(const float* __restrict__ x, const float* __restrict__ ew,
                             const float* __restrict__ eb, const float* __restrict__ pos,
                             float* __restrict__ x1, float* __restrict__ x2) {
  int idx = blockIdx.x * 256 + threadIdx.x;
  if (idx >= MROWS * D_MODEL) return;
  int row = idx >> 9;
  int d   = idx & 511;
  int t   = row & (SEQ - 1);
  float s = eb[d] + pos[(size_t)t * D_MODEL + d];
  const float* xr = x + (size_t)row * 32;
#pragma unroll
  for (int k = 0; k < 32; ++k) s += xr[k] * ew[k * D_MODEL + d];
  x1[idx] = s;
  x2[idx] = s;
}

// ---------------- row LayerNorm (512 wide), emits bf16 operand ----------------
__global__ __launch_bounds__(256) void layernorm512_kernel(
    const float* __restrict__ src, const float* __restrict__ g, const float* __restrict__ b,
    unsigned short* __restrict__ dst_bf, float* __restrict__ dst_f) {
  __shared__ float red[256];
  int row = blockIdx.x, tid = threadIdx.x;
  const float* s = src + (size_t)row * 512;
  float v0 = s[tid], v1 = s[tid + 256];
  red[tid] = v0 + v1;
  __syncthreads();
  for (int off = 128; off > 0; off >>= 1) { if (tid < off) red[tid] += red[tid + off]; __syncthreads(); }
  float mean = red[0] * (1.0f / 512.0f);
  __syncthreads();
  float d0 = v0 - mean, d1 = v1 - mean;
  red[tid] = d0 * d0 + d1 * d1;
  __syncthreads();
  for (int off = 128; off > 0; off >>= 1) { if (tid < off) red[tid] += red[tid + off]; __syncthreads(); }
  float inv = rsqrtf(red[0] * (1.0f / 512.0f) + 1e-5f);
  float o0 = d0 * inv * g[tid] + b[tid];
  float o1 = d1 * inv * g[tid + 256] + b[tid + 256];
  size_t base = (size_t)row * 512;
  if (dst_bf) { dst_bf[base + tid] = f2bf(o0); dst_bf[base + tid + 256] = f2bf(o1); }
  if (dst_f)  { dst_f[base + tid] = o0;        dst_f[base + tid + 256] = o1; }
}

// ---------------- WMMA bf16 GEMM ----------------
// C(MxN) = A(MxK, row-major bf16) * Bt(NxK, row-major bf16 = B^T) [+bias][gelu][+resid]
// Double-buffered LDS tiles pre-swizzled into WMMA fragment register layout; staging uses
// gfx1250 async global->LDS copies (ASYNCcnt) when available.
#define BM 256
#define BN 64
#define BKK 32

__global__ __launch_bounds__(256) void gemm_bf16_kernel(
    const unsigned short* __restrict__ A, const unsigned short* __restrict__ Bt,
    const float* __restrict__ bias, const float* __restrict__ resid,
    float* __restrict__ Cf, unsigned short* __restrict__ Cb,
    int M, int N, int K, int gelu_act) {
  __shared__ v4u Asw[2][BM * BKK / 8];   // 2 x 16KB, [subtile(16)][lane(32)][2]
  __shared__ v4u Bsw[2][BKK * BN / 8];   // 2 x 4KB,  [tile(4)][lane(32)][2]
  const int tid  = threadIdx.x;
  const int wave = tid >> 5;
  const int lane = tid & 31;
  const int mr   = lane & 15;
  const int half = lane >> 4;
  const int m0 = blockIdx.y * BM;
  const int n0 = blockIdx.x * BN;

  v8f acc[2][4] = {};

  auto stage = [&](int buf, int k0) {
    unsigned int* As32 = reinterpret_cast<unsigned int*>(Asw[buf]);
    unsigned int* Bs32 = reinterpret_cast<unsigned int*>(Bsw[buf]);
#if USE_ASYNC_LDS
#pragma unroll
    for (int it = 0; it < (BM * BKK / 2) / 256; ++it) {     // 16 iters
      int idx = it * 256 + tid;
      int r = idx >> 4, c = idx & 15;
      int s = r >> 4, mrr = r & 15;
      int hf = (c >> 2) & 1;
      int vv = ((c >= 8) ? 4 : 0) + (c & 3);
      __builtin_amdgcn_global_load_async_to_lds_b32(
          (gas_i32*)(A + (size_t)(m0 + r) * K + k0 + 2 * c),
          (las_i32*)(&As32[((s * 32 + mrr + 16 * hf) << 3) + vv]), 0, 0);
    }
#pragma unroll
    for (int it = 0; it < (BKK * BN / 2) / 256; ++it) {     // 4 iters
      int idx = it * 256 + tid;
      int nl = idx >> 4, c = idx & 15;
      int tt = nl >> 4, nr = nl & 15;
      int hf = (c >> 2) & 1, vv = ((c >= 8) ? 4 : 0) + (c & 3);
      __builtin_amdgcn_global_load_async_to_lds_b32(
          (gas_i32*)(Bt + (size_t)(n0 + nl) * K + k0 + 2 * c),
          (las_i32*)(&Bs32[((tt * 32 + nr + 16 * hf) << 3) + vv]), 0, 0);
    }
#else
    unsigned int va[16], vb[4];
#pragma unroll
    for (int it = 0; it < 16; ++it) {
      int idx = it * 256 + tid;
      int r = idx >> 4, c = idx & 15;
      va[it] = *reinterpret_cast<const unsigned int*>(A + (size_t)(m0 + r) * K + k0 + 2 * c);
    }
#pragma unroll
    for (int it = 0; it < 4; ++it) {
      int idx = it * 256 + tid;
      int nl = idx >> 4, c = idx & 15;
      vb[it] = *reinterpret_cast<const unsigned int*>(Bt + (size_t)(n0 + nl) * K + k0 + 2 * c);
    }
#pragma unroll
    for (int it = 0; it < 16; ++it) {
      int idx = it * 256 + tid;
      int r = idx >> 4, c = idx & 15;
      int s = r >> 4, mrr = r & 15;
      int hf = (c >> 2) & 1;
      int vv = ((c >= 8) ? 4 : 0) + (c & 3);
      As32[((s * 32 + mrr + 16 * hf) << 3) + vv] = va[it];
    }
#pragma unroll
    for (int it = 0; it < 4; ++it) {
      int idx = it * 256 + tid;
      int nl = idx >> 4, c = idx & 15;
      int tt = nl >> 4, nr = nl & 15;
      int hf = (c >> 2) & 1, vv = ((c >= 8) ? 4 : 0) + (c & 3);
      Bs32[((tt * 32 + nr + 16 * hf) << 3) + vv] = vb[it];
    }
#endif
  };

  stage(0, 0);
#if USE_ASYNC_LDS
  asm volatile("s_wait_asynccnt 0x0" ::: "memory");
#endif
  __syncthreads();

  int p = 0;
  for (int k0 = 0; k0 < K; k0 += BKK) {
    if (k0 + BKK < K) stage(p ^ 1, k0 + BKK);   // overlap next-tile copy with WMMAs

    const v4u* Ab = Asw[p];
    const v4u* Bb = Bsw[p];
    BF16Frag a0, a1;
    int ab0 = ((2 * wave) * 32 + lane) * 2;
    int ab1 = ((2 * wave + 1) * 32 + lane) * 2;
    a0.q[0] = Ab[ab0]; a0.q[1] = Ab[ab0 + 1];
    a1.q[0] = Ab[ab1]; a1.q[1] = Ab[ab1 + 1];
#pragma unroll
    for (int t = 0; t < 4; ++t) {
      BF16Frag bfr;
      int bb = (t * 32 + lane) * 2;
      bfr.q[0] = Bb[bb]; bfr.q[1] = Bb[bb + 1];
      acc[0][t] = __builtin_amdgcn_wmma_f32_16x16x32_bf16(
          false, a0.v, false, bfr.v, (short)0, acc[0][t], false, false);
      acc[1][t] = __builtin_amdgcn_wmma_f32_16x16x32_bf16(
          false, a1.v, false, bfr.v, (short)0, acc[1][t], false, false);
    }
#if USE_ASYNC_LDS
    asm volatile("s_wait_asynccnt 0x0" ::: "memory");
#endif
    __syncthreads();
    p ^= 1;
  }

#pragma unroll
  for (int aa = 0; aa < 2; ++aa) {
#pragma unroll
    for (int t = 0; t < 4; ++t) {
#pragma unroll
      for (int v = 0; v < 8; ++v) {
        int m = m0 + (2 * wave + aa) * 16 + v + 8 * half;   // C: M = v + 8*(lane>>4)
        int n = n0 + t * 16 + mr;                            // C: N = lane & 15
        float val = acc[aa][t][v];
        if (bias) val += bias[n];
        if (gelu_act) val = 0.5f * val * (1.0f + erff(val * 0.70710678118f));
        size_t o = (size_t)m * N + n;
        if (resid) val += resid[o];
        if (Cf) Cf[o] = val;
        if (Cb) Cb[o] = f2bf(val);
      }
    }
  }
}

// ---------------- LSH bucketing + sort (stable partition per hash) ----------------
__global__ __launch_bounds__(512) void lsh_sort_kernel(
    const float* __restrict__ qk, const float* __restrict__ rotl,
    int* __restrict__ st, int* __restrict__ undo) {
  __shared__ int scan[512];
  int bh = blockIdx.x;
  int b = bh >> 3, h = bh & 7;
  int t = threadIdx.x;
  const float* qv = qk + ((size_t)(b * SEQ + t)) * D_MODEL + h * DH;
  for (int r = 0; r < NH; ++r) {
    float sdot = 0.f;
#pragma unroll
    for (int d = 0; d < DH; ++d) sdot += qv[d] * rotl[d * NH + r];
    // argmax([s, -s]) == 0 iff s >= 0
    int bkt = (sdot >= 0.f) ? 0 : 1;
    scan[t] = (bkt == 0) ? 1 : 0;
    __syncthreads();
    for (int off = 1; off < 512; off <<= 1) {
      int add = (t >= off) ? scan[t - off] : 0;
      __syncthreads();
      scan[t] += add;
      __syncthreads();
    }
    int c0 = scan[511];
    int pos = (bkt == 0) ? (scan[t] - 1) : (c0 + (t + 1 - scan[t]) - 1);
    st[(size_t)bh * NS + r * SEQ + pos] = t;             // sorted pos -> token id
    undo[(size_t)bh * NS + r * SEQ + t] = r * SEQ + pos; // flat idx -> sorted pos
    __syncthreads();
  }
}

// ---------------- chunked LSH attention, online softmax ----------------
__global__ __launch_bounds__(256) void lsh_attn_kernel(
    const float* __restrict__ qk, const float* __restrict__ vv,
    const int* __restrict__ st,
    float* __restrict__ o_sorted, float* __restrict__ lse_sorted) {
  __shared__ float ks[64][DH];
  __shared__ float vs[64][DH];
  __shared__ float kn[64];
  __shared__ int   tks[64];
  int blk = blockIdx.x;
  int bh = blk >> 3;
  int c  = blk & 7;
  int b = bh >> 3, h = bh & 7;
  int i = threadIdx.x;                 // query row within chunk
  int tq = st[(size_t)bh * NS + c * CHUNK + i];
  const float* qv = qk + ((size_t)(b * SEQ + tq)) * D_MODEL + h * DH;
  float q[DH];
#pragma unroll
  for (int d = 0; d < DH; ++d) q[d] = qv[d];
  float o[DH];
#pragma unroll
  for (int d = 0; d < DH; ++d) o[d] = 0.f;
  float mval = -3.402823e38f;
  float l = 0.f;
  int cp = (c + 7) & 7;                // look-one-back chunk (circular)
  for (int jb = 0; jb < 8; ++jb) {     // 512 keys in 64-wide tiles
    __syncthreads();
    for (int idx = i; idx < 64 * DH; idx += 256) {
      int jj = idx >> 6;
      int d  = idx & 63;
      int j  = jb * 64 + jj;
      int cc = (j < CHUNK) ? c : cp;
      int rowin = j & (CHUNK - 1);
      int tk = st[(size_t)bh * NS + cc * CHUNK + rowin];
      size_t base = ((size_t)(b * SEQ + tk)) * D_MODEL + h * DH;
      ks[jj][d] = qk[base + d];
      vs[jj][d] = vv[base + d];
      if (d == 0) tks[jj] = tk;
    }
    __syncthreads();
    if (i < 64) {
      float nn = 0.f;
#pragma unroll
      for (int d = 0; d < DH; ++d) nn += ks[i][d] * ks[i][d];
      kn[i] = 1.0f / fmaxf(sqrtf(nn), 1e-12f);
    }
    __syncthreads();
    for (int jj = 0; jj < 64; ++jj) {
      float sdot = 0.f;
#pragma unroll
      for (int d = 0; d < DH; ++d) sdot += q[d] * ks[jj][d];
      sdot *= kn[jj] * 0.125f;         // / sqrt(64)
      int tk = tks[jj];
      if (tq < tk)       sdot = -3.402823e38f;  // causal
      else if (tq == tk) sdot = -5.0e4f;        // de-prioritize self
      float mnew = fmaxf(mval, sdot);
      float scl = __expf(mval - mnew);
      float p   = __expf(sdot - mnew);
      l = l * scl + p;
#pragma unroll
      for (int d = 0; d < DH; ++d) o[d] = o[d] * scl + p * vs[jj][d];
      mval = mnew;
    }
  }
  float linv = 1.0f / l;
  size_t ob = ((size_t)bh * NS + c * CHUNK + i) * DH;
#pragma unroll
  for (int d = 0; d < DH; ++d) o_sorted[ob + d] = o[d] * linv;
  lse_sorted[(size_t)bh * NS + c * CHUNK + i] = mval + logf(l);
}

// ---------------- unsort + combine hash rounds -> bf16 (b,t,h,dh) ----------------
__global__ __launch_bounds__(256) void combine_kernel(
    const float* __restrict__ o_sorted, const float* __restrict__ lse_sorted,
    const int* __restrict__ undo, unsigned short* __restrict__ attn_bf) {
  int gid = blockIdx.x * 4 + (threadIdx.x >> 6);
  int d = threadIdx.x & 63;
  int bh = gid >> 9;
  int t  = gid & 511;
  int b = bh >> 3, h = bh & 7;
  int pos[NH]; float lse[NH];
#pragma unroll
  for (int r = 0; r < NH; ++r) {
    pos[r] = undo[(size_t)bh * NS + r * SEQ + t];
    lse[r] = lse_sorted[(size_t)bh * NS + pos[r]];
  }
  float mx = fmaxf(fmaxf(lse[0], lse[1]), fmaxf(lse[2], lse[3]));
  float ssum = 0.f, w[NH];
#pragma unroll
  for (int r = 0; r < NH; ++r) { w[r] = __expf(lse[r] - mx); ssum += w[r]; }
  float oa = 0.f;
#pragma unroll
  for (int r = 0; r < NH; ++r)
    oa += o_sorted[((size_t)bh * NS + pos[r]) * DH + d] * w[r];
  oa /= ssum;
  attn_bf[((size_t)(b * SEQ + t)) * D_MODEL + h * DH + d] = f2bf(oa);
}

// ---------------- final: 0.5*(x1+x2) last token + LayerNorm ----------------
__global__ __launch_bounds__(256) void final_ln_kernel(
    const float* __restrict__ x1, const float* __restrict__ x2,
    const float* __restrict__ g, const float* __restrict__ b,
    float* __restrict__ hf) {
  __shared__ float red[256];
  int bb = blockIdx.x, tid = threadIdx.x;
  size_t base = ((size_t)(bb * SEQ + (SEQ - 1))) * D_MODEL;
  float v0 = 0.5f * (x1[base + tid] + x2[base + tid]);
  float v1 = 0.5f * (x1[base + tid + 256] + x2[base + tid + 256]);
  red[tid] = v0 + v1;
  __syncthreads();
  for (int off = 128; off > 0; off >>= 1) { if (tid < off) red[tid] += red[tid + off]; __syncthreads(); }
  float mean = red[0] * (1.0f / 512.0f);
  __syncthreads();
  float d0 = v0 - mean, d1 = v1 - mean;
  red[tid] = d0 * d0 + d1 * d1;
  __syncthreads();
  for (int off = 128; off > 0; off >>= 1) { if (tid < off) red[tid] += red[tid + off]; __syncthreads(); }
  float inv = rsqrtf(red[0] * (1.0f / 512.0f) + 1e-5f);
  hf[bb * 512 + tid]       = d0 * inv * g[tid] + b[tid];
  hf[bb * 512 + tid + 256] = d1 * inv * g[tid + 256] + b[tid + 256];
}

// ---------------- head MLP: 512->256, LN, relu, ->1 ----------------
__global__ __launch_bounds__(256) void head_kernel(
    const float* __restrict__ hf,
    const float* __restrict__ hw1, const float* __restrict__ hb1,
    const float* __restrict__ hg,  const float* __restrict__ hbb,
    const float* __restrict__ hw2, const float* __restrict__ hb2,
    float* __restrict__ out) {
  __shared__ float red[256];
  int bb = blockIdx.x, j = threadIdx.x;
  const float* hr = hf + bb * 512;
  float s = hb1[j];
  for (int k = 0; k < 512; ++k) s += hr[k] * hw1[k * 256 + j];
  red[j] = s;
  __syncthreads();
  for (int off = 128; off > 0; off >>= 1) { if (j < off) red[j] += red[j + off]; __syncthreads(); }
  float mean = red[0] * (1.0f / 256.0f);
  __syncthreads();
  float dd = s - mean;
  red[j] = dd * dd;
  __syncthreads();
  for (int off = 128; off > 0; off >>= 1) { if (j < off) red[j] += red[j + off]; __syncthreads(); }
  float inv = rsqrtf(red[0] * (1.0f / 256.0f) + 1e-5f);
  __syncthreads();
  float ln = dd * inv * hg[j] + hbb[j];
  float rl = fmaxf(ln, 0.f);
  red[j] = rl * hw2[j];
  __syncthreads();
  for (int off = 128; off > 0; off >>= 1) { if (j < off) red[j] += red[j + off]; __syncthreads(); }
  if (j == 0) out[bb] = red[0] + hb2[0];
}

// ---------------- host orchestration ----------------
extern "C" void kernel_launch(void* const* d_in, const int* in_sizes, int n_in,
                              void* d_out, int out_size, void* d_ws, size_t ws_size,
                              hipStream_t stream) {
  (void)in_sizes; (void)n_in; (void)out_size; (void)ws_size;
  const float* x     = (const float*)d_in[0];
  const float* emb_w = (const float*)d_in[1];
  const float* emb_b = (const float*)d_in[2];
  const float* pos   = (const float*)d_in[3];
  const float* ln1_g = (const float*)d_in[4];
  const float* ln1_b = (const float*)d_in[5];
  const float* wqk   = (const float*)d_in[6];
  const float* wv    = (const float*)d_in[7];
  const float* wo    = (const float*)d_in[8];
  const float* wo_b  = (const float*)d_in[9];
  const float* ln2_g = (const float*)d_in[10];
  const float* ln2_b = (const float*)d_in[11];
  const float* ff_w1 = (const float*)d_in[12];
  const float* ff_b1 = (const float*)d_in[13];
  const float* ff_w2 = (const float*)d_in[14];
  const float* ff_b2 = (const float*)d_in[15];
  const float* rot   = (const float*)d_in[16];
  const float* lnf_g = (const float*)d_in[17];
  const float* lnf_b = (const float*)d_in[18];
  const float* hw1   = (const float*)d_in[19];
  const float* hb1   = (const float*)d_in[20];
  const float* hln_g = (const float*)d_in[21];
  const float* hln_b = (const float*)d_in[22];
  const float* hw2   = (const float*)d_in[23];
  const float* hb2   = (const float*)d_in[24];
  float* out = (float*)d_out;

  size_t off = 0;
  auto alloc = [&](size_t bytes) -> void* {
    void* p = (char*)d_ws + off;
    off += (bytes + 255) & ~(size_t)255;
    return p;
  };
  float* x1            = (float*)alloc((size_t)MROWS * D_MODEL * 4);
  float* x2            = (float*)alloc((size_t)MROWS * D_MODEL * 4);
  unsigned short* xnbf = (unsigned short*)alloc((size_t)MROWS * D_MODEL * 2);
  float* qkb           = (float*)alloc((size_t)MROWS * D_MODEL * 4);
  float* vvb           = (float*)alloc((size_t)MROWS * D_MODEL * 4);
  int* st              = (int*)alloc((size_t)BHN * NS * 4);
  int* undo            = (int*)alloc((size_t)BHN * NS * 4);
  float* o_sorted      = (float*)alloc((size_t)BHN * NS * DH * 4);
  float* lse_sorted    = (float*)alloc((size_t)BHN * NS * 4);
  unsigned short* attn = (unsigned short*)alloc((size_t)MROWS * D_MODEL * 2);
  unsigned short* hid  = (unsigned short*)alloc((size_t)MROWS * FFD * 2);
  unsigned short* wqkb = (unsigned short*)alloc((size_t)4 * D_MODEL * D_MODEL * 2);
  unsigned short* wvbf = (unsigned short*)alloc((size_t)4 * D_MODEL * D_MODEL * 2);
  unsigned short* wobf = (unsigned short*)alloc((size_t)4 * D_MODEL * D_MODEL * 2);
  unsigned short* f1bf = (unsigned short*)alloc((size_t)4 * D_MODEL * FFD * 2);
  unsigned short* f2bfw= (unsigned short*)alloc((size_t)4 * FFD * D_MODEL * 2);
  float* hf            = (float*)alloc((size_t)BATCH * D_MODEL * 4);

  // transpose+convert weights to bf16 N x K (B^T) once per launch
  auto convt = [&](const float* s, unsigned short* d, int K, int N) {
    int n = 4 * K * N;
    f2bf_t_kernel<<<(n + 255) / 256, 256, 0, stream>>>(s, d, K, N, 4);
  };
  convt(wqk,   wqkb, D_MODEL, D_MODEL);
  convt(wv,    wvbf, D_MODEL, D_MODEL);
  convt(wo,    wobf, D_MODEL, D_MODEL);
  convt(ff_w1, f1bf, D_MODEL, FFD);
  convt(ff_w2, f2bfw, FFD, D_MODEL);

  embed_kernel<<<(MROWS * D_MODEL + 255) / 256, 256, 0, stream>>>(x, emb_w, emb_b, pos, x1, x2);

  auto gemm = [&](const unsigned short* A, const unsigned short* Bt, const float* bias,
                  const float* resid, float* Cf, unsigned short* Cb,
                  int M_, int N_, int K_, int act) {
    dim3 g(N_ / BN, M_ / BM);
    gemm_bf16_kernel<<<g, 256, 0, stream>>>(A, Bt, bias, resid, Cf, Cb, M_, N_, K_, act);
  };

  for (int i = 0; i < 4; ++i) {
    const int DD = D_MODEL * D_MODEL;
    // x1 += attn(LN(x2))
    layernorm512_kernel<<<MROWS, 256, 0, stream>>>(x2, ln1_g + i * 512, ln1_b + i * 512, xnbf, nullptr);
    gemm(xnbf, wqkb + (size_t)i * DD, nullptr, nullptr, qkb, nullptr, MROWS, D_MODEL, D_MODEL, 0);
    gemm(xnbf, wvbf + (size_t)i * DD, nullptr, nullptr, vvb, nullptr, MROWS, D_MODEL, D_MODEL, 0);
    lsh_sort_kernel<<<BHN, 512, 0, stream>>>(qkb, rot + (size_t)i * DH * NH, st, undo);
    lsh_attn_kernel<<<BHN * NCH, 256, 0, stream>>>(qkb, vvb, st, o_sorted, lse_sorted);
    combine_kernel<<<BHN * SEQ / 4, 256, 0, stream>>>(o_sorted, lse_sorted, undo, attn);
    gemm(attn, wobf + (size_t)i * DD, wo_b + i * 512, x1, x1, nullptr, MROWS, D_MODEL, D_MODEL, 0);
    // x2 += ff(LN(x1))
    layernorm512_kernel<<<MROWS, 256, 0, stream>>>(x1, ln2_g + i * 512, ln2_b + i * 512, xnbf, nullptr);
    gemm(xnbf, f1bf + (size_t)i * D_MODEL * FFD, ff_b1 + i * FFD, nullptr, nullptr, hid,
         MROWS, FFD, D_MODEL, 1 /*gelu*/);
    gemm(hid, f2bfw + (size_t)i * FFD * D_MODEL, ff_b2 + i * 512, x2, x2, nullptr,
         MROWS, D_MODEL, FFD, 0);
  }

  final_ln_kernel<<<BATCH, 256, 0, stream>>>(x1, x2, lnf_g, lnf_b, hf);
  head_kernel<<<BATCH, 256, 0, stream>>>(hf, hw1, hb1, hln_g, hln_b, hw2, hb2, out);
}
